// XARNN_25005299597677
// MI455X (gfx1250) — compile-verified, matrix operand order checked by reading, not run on hardware
//
#include <hip/hip_runtime.h>
#include <hip/hip_bf16.h>
#include <stdint.h>

// Problem sizes (compile-time constants from the reference)
#define B_SZ 32
#define T_SZ 2048
#define D_IN 256
#define H_SZ 512

typedef _Float16 v8h  __attribute__((ext_vector_type(8)));
typedef _Float16 v16h __attribute__((ext_vector_type(16)));
typedef float    v8f  __attribute__((ext_vector_type(8)));
typedef float    v4f  __attribute__((ext_vector_type(4)));
typedef unsigned int u32x4 __attribute__((ext_vector_type(4)));
typedef int          i32x4 __attribute__((ext_vector_type(4)));
typedef int          i32x8 __attribute__((ext_vector_type(8)));

// ---- workspace layout (bytes) ----
// whh16: 32 coltiles x 16 kchunks x (32 lanes x 16 f16)  = 512 KB  (WMMA B fragments)
// wxh16: 32 coltiles x  8 kchunks x (32 lanes x 16 f16)  = 256 KB
// xw16 : 2048*2*32 C-fragments x (32 lanes x 8 f16)      =  64 MB  (xW + bh, pre-swizzled)
#define WS_WHH16   0u
#define WS_WXH16   (512u*1024u)
#define WS_XW16    (768u*1024u)
#define WS_QP      (WS_XW16 + 64u*1024u*1024u)   // 32x512 f32
#define WS_SCORES  (WS_QP + 64u*1024u)           // 32x2048 f32
#define WS_ATT     (WS_SCORES + 256u*1024u)      // 32x2048 f32
#define WS_CPART   (WS_ATT + 256u*1024u)         // 8x32x512 f32 partial contexts

#if defined(__has_builtin)
# if __has_builtin(__builtin_amdgcn_tensor_load_to_lds)
#  define HAS_TDM 1
# endif
# if __has_builtin(__builtin_amdgcn_s_wait_tensorcnt)
#  define HAS_TWAIT 1
# endif
#endif

__device__ __forceinline__ float fast_tanh(float x){
#if defined(__has_builtin)
# if __has_builtin(__builtin_amdgcn_tanhf)
  return __builtin_amdgcn_tanhf(x);   // V_TANH_F32 (gfx1250 TRANS op)
# else
  return tanhf(x);
# endif
#else
  return tanhf(x);
#endif
}

__device__ __forceinline__ v16h cat16(v8h lo, v8h hi){
  return __builtin_shufflevector(lo, hi, 0,1,2,3,4,5,6,7,8,9,10,11,12,13,14,15);
}

#define WMMA_F16(A,Bm,C) __builtin_amdgcn_wmma_f32_16x16x32_f16(false,(A),false,(Bm),(short)0,(C),false,false)

// TDM: load a 2D tile (tile0 units of 8B per row, tile1 rows, row stride stride0 units)
// from global ga into LDS at byte-offset lds_off. D# built per ISA 08_async_tensor §8.
__device__ __forceinline__ void tdm_load_2d(unsigned lds_off, unsigned long long ga,
                                            unsigned tile0, unsigned tile1,
                                            unsigned long long stride0,
                                            unsigned td0, unsigned td1){
#ifdef HAS_TDM
  u32x4 g0;
  g0[0] = 1u;                                                    // count=1 (user D#)
  g0[1] = lds_off;                                               // lds_addr
  g0[2] = (unsigned)(ga & 0xffffffffull);                        // global_addr[31:0]
  g0[3] = ((unsigned)(ga >> 32) & 0x01ffffffu) | (2u << 30);     // addr[56:32] | type=2
  i32x8 g1;
  g1[0] = (int)(3u << 16);                                       // data_size=8B
  g1[1] = (int)((td0 & 0xffffu) << 16);                          // tensor_dim0[15:0]
  g1[2] = (int)(((td0 >> 16) & 0xffffu) | ((td1 & 0xffffu) << 16));
  g1[3] = (int)(((td1 >> 16) & 0xffffu) | ((tile0 & 0xffffu) << 16));
  g1[4] = (int)(tile1 & 0xffffu);                                // tile_dim1 (tile_dim2=0)
  g1[5] = (int)(unsigned)(stride0 & 0xffffffffull);              // dim0_stride lo32
  g1[6] = (int)(unsigned)((stride0 >> 32) & 0xffffull);          // dim0_stride hi16
  g1[7] = 0;
  i32x4 g2 = {0,0,0,0};
  i32x4 g3 = {0,0,0,0};
# if __clang_major__ >= 23
  i32x8 g4 = {0,0,0,0,0,0,0,0};
  __builtin_amdgcn_tensor_load_to_lds(g0, g1, g2, g3, g4, 0);
# else
  __builtin_amdgcn_tensor_load_to_lds(g0, g1, g2, g3, 0);
# endif
#else
  (void)lds_off; (void)ga; (void)tile0; (void)tile1; (void)stride0; (void)td0; (void)td1;
#endif
}

__device__ __forceinline__ void tdm_wait(){
#ifdef HAS_TWAIT
  __builtin_amdgcn_s_wait_tensorcnt(0);
#endif
}

// K index inside a 16-bit 16x32 A/B fragment (ISA 05_wmma §7.12.2)
__device__ __forceinline__ int kidx16(int j, int half, int p){
  return ((j >= 4) ? 16 : 0) + half*8 + ((j & 3) << 1) + p;
}

// ---------------------------------------------------------------------------
// Kernel 1: convert Whh/Wxh f32 -> f16 WMMA B-fragments in ws
// ---------------------------------------------------------------------------
__global__ void k_prep(const float* __restrict__ Wxh, const float* __restrict__ Whh,
                       _Float16* __restrict__ whh16, _Float16* __restrict__ wxh16){
  int i = blockIdx.x*256 + threadIdx.x;
  if (i < 262144){                                   // 512*512 Whh elements
    int p = i & 1, j = (i>>1)&7, lane = (i>>4)&31, kc = (i>>9)&15, c = i>>13;
    int n = c*16 + (lane & 15), half = lane >> 4;
    int k = kc*32 + kidx16(j, half, p);
    whh16[(((c*16 + kc)*32 + lane) << 4) + j*2 + p] = (_Float16)Whh[k*H_SZ + n];
  } else {
    int ii = i - 262144;
    if (ii < 131072){                                // 256*512 Wxh elements
      int p = ii & 1, j = (ii>>1)&7, lane = (ii>>4)&31, kc = (ii>>9)&7, c = ii>>12;
      int n = c*16 + (lane & 15), half = lane >> 4;
      int k = kc*32 + kidx16(j, half, p);
      wxh16[(((c*8 + kc)*32 + lane) << 4) + j*2 + p] = (_Float16)Wxh[k*H_SZ + n];
    }
  }
}

// ---------------------------------------------------------------------------
// Kernel 2: xW[t,b,h] = X@Wxh + bh, stored as f16 WMMA C-fragments.
// 16 waves/WG, one (t, batch-half) strip per wave, Wxh fragments staged in LDS via TDM.
// ---------------------------------------------------------------------------
__global__ __launch_bounds__(512) void k_xw(const float* __restrict__ X,
                                            const float* __restrict__ bh,
                                            const _Float16* __restrict__ wxh16,
                                            _Float16* __restrict__ xw16){
  extern __shared__ char dynsmem[];
  _Float16* bfrag = (_Float16*)dynsmem;     // 256 KB: 32 tiles x 8 chunks x 1KB
#ifdef HAS_TDM
  if (threadIdx.x < 32){
    tdm_load_2d((unsigned)(size_t)bfrag, (unsigned long long)(size_t)wxh16,
                32768u, 1u, 32768ull, 32768u, 1u);          // 1D 256KB (8B units)
    tdm_wait();
  }
#else
  for (int i = threadIdx.x; i < (256*1024/16); i += 512)
    ((uint4*)bfrag)[i] = ((const uint4*)wxh16)[i];
#endif
  __syncthreads();

  int wave = threadIdx.x >> 5, lane = threadIdx.x & 31;
  int strip = blockIdx.x*16 + wave;          // 4096 strips = 2048 t x 2 batch halves
  int t = strip >> 1, g = strip & 1;
  int row = lane & 15, half = lane >> 4;
  const float* xrow = X + ((size_t)(g*16 + row)*T_SZ + t)*D_IN;

  v16h a[8];                                 // A: 16x256 f16 (this wave's 16 X rows)
  #pragma unroll
  for (int kc = 0; kc < 8; ++kc){
    #pragma unroll
    for (int j = 0; j < 8; ++j){
      int d0 = kc*32 + kidx16(j, half, 0);
      float2 xv = *(const float2*)(xrow + d0);
      a[kc][2*j]   = (_Float16)xv.x;
      a[kc][2*j+1] = (_Float16)xv.y;
    }
  }
  for (int c = 0; c < 32; ++c){
    v8f acc = {0,0,0,0,0,0,0,0};
    #pragma unroll
    for (int kc = 0; kc < 8; ++kc){
      const _Float16* bp = bfrag + (((c*8 + kc)*32 + lane) << 4);
      v16h bm = cat16(*(const v8h*)bp, *(const v8h*)(bp + 8));
      acc = WMMA_F16(a[kc], bm, acc);
    }
    float bias = bh[c*16 + row];             // bake bias into xW
    v8h out;
    #pragma unroll
    for (int i = 0; i < 8; ++i) out[i] = (_Float16)(acc[i] + bias);
    *(v8h*)(xw16 + ((((size_t)t*2 + g)*32 + c)*32 + lane)*8) = out;
  }
}

// ---------------------------------------------------------------------------
// Kernel 3: sequential scan. 2 WGs (one per 16-row batch half), 16 waves each.
// h double-buffered in LDS (f16). Whh chunks 0..7 VGPR-resident, 8..15 in LDS (TDM-staged).
// ---------------------------------------------------------------------------
__global__ __launch_bounds__(512) void k_scan(const _Float16* __restrict__ whh16,
                                              const _Float16* __restrict__ xw16,
                                              float* __restrict__ Yout){
  __shared__ _Float16 hbuf[2][16*H_SZ];      // 32 KB double-buffered hidden state
  extern __shared__ char dynsmem[];
  _Float16* whihi = (_Float16*)dynsmem;      // 256 KB: chunks 8..15 for all 32 col tiles

  int g = blockIdx.x;
  int wave = threadIdx.x >> 5, lane = threadIdx.x & 31;
  int row = lane & 15, half = lane >> 4;

#ifdef HAS_TDM
  if (threadIdx.x < 32){
    // 32 rows of 8KB (chunks 8..15 per col tile), source row stride 16KB
    tdm_load_2d((unsigned)(size_t)whihi,
                (unsigned long long)(size_t)(whh16 + 8*512),
                1024u, 32u, 2048ull, 2048u, 32u);
    tdm_wait();
  }
#else
  for (int i = threadIdx.x; i < (256*1024/16); i += 512){
    int c = i >> 9, r = i & 511;
    ((uint4*)whihi)[i] = *(const uint4*)((const char*)whh16 + (c*16 + 8)*1024 + r*16);
  }
#endif

  // VGPR-resident B fragments: K-chunks 0..7 for col tiles c0=wave, c1=wave+16
  int c0 = wave, c1 = wave + 16;
  v16h bwa[2][8];
  #pragma unroll
  for (int e = 0; e < 2; ++e){
    int c = wave + e*16;
    #pragma unroll
    for (int kc = 0; kc < 8; ++kc){
      const _Float16* bp = whh16 + (((c*16 + kc)*32 + lane) << 4);
      bwa[e][kc] = cat16(*(const v8h*)bp, *(const v8h*)(bp + 8));
    }
  }
  // h(0) = 0
  for (int i = threadIdx.x; i < (16*H_SZ/8); i += 512)
    ((uint4*)&hbuf[0][0])[i] = make_uint4(0u,0u,0u,0u);
  __syncthreads();

  v8h xw0 = *(const v8h*)(xw16 + (((size_t)g*32 + c0)*32 + lane)*8);
  v8h xw1 = *(const v8h*)(xw16 + (((size_t)g*32 + c1)*32 + lane)*8);

  for (int t = 0; t < T_SZ; ++t){
    int p = t & 1;
    v8f acc0, acc1;
    #pragma unroll
    for (int i = 0; i < 8; ++i){ acc0[i] = (float)xw0[i]; acc1[i] = (float)xw1[i]; }

    // prefetch next step's xW fragments (hide global latency behind WMMAs)
    v8h nxw0 = xw0, nxw1 = xw1;
    if (t + 1 < T_SZ){
      nxw0 = *(const v8h*)(xw16 + ((((size_t)(t+1)*2 + g)*32 + c0)*32 + lane)*8);
      nxw1 = *(const v8h*)(xw16 + ((((size_t)(t+1)*2 + g)*32 + c1)*32 + lane)*8);
    }

    const _Float16* hb = &hbuf[p][0];
    #pragma unroll
    for (int kc = 0; kc < 16; ++kc){
      const _Float16* ap = hb + row*H_SZ + kc*32 + half*8;
      v16h amat = cat16(*(const v8h*)ap, *(const v8h*)(ap + 16));
      if (kc < 8){
        acc0 = WMMA_F16(amat, bwa[0][kc], acc0);
        acc1 = WMMA_F16(amat, bwa[1][kc], acc1);
      } else {
        const _Float16* b0p = whihi + (((c0*8 + (kc-8))*32 + lane) << 4);
        const _Float16* b1p = whihi + (((c1*8 + (kc-8))*32 + lane) << 4);
        v16h b0 = cat16(*(const v8h*)b0p, *(const v8h*)(b0p + 8));
        v16h b1 = cat16(*(const v8h*)b1p, *(const v8h*)(b1p + 8));
        acc0 = WMMA_F16(amat, b0, acc0);
        acc1 = WMMA_F16(amat, b1, acc1);
      }
    }

    _Float16* hn = &hbuf[1 - p][0];
    #pragma unroll
    for (int r = 0; r < 8; ++r){
      int m = r + half*8;                            // C layout: lanes16-31 -> M=8+r
      float y0 = fast_tanh(acc0[r]);
      float y1 = fast_tanh(acc1[r]);
      size_t yo = ((size_t)t*B_SZ + g*16 + m)*H_SZ;
      Yout[yo + c0*16 + row] = y0;
      Yout[yo + c1*16 + row] = y1;
      hn[m*H_SZ + c0*16 + row] = (_Float16)y0;
      hn[m*H_SZ + c1*16 + row] = (_Float16)y1;
    }
    __syncthreads();
    xw0 = nxw0; xw1 = nxw1;
  }
}

// ---------------------------------------------------------------------------
// Attention (K never materialized): qp[b,:] = KW @ (Y_last[b,:] @ QW)
// ---------------------------------------------------------------------------
__global__ void k_q0qp(const float* __restrict__ Y, const float* __restrict__ QW,
                       const float* __restrict__ KW, float* __restrict__ qp){
  __shared__ float q0[B_SZ*D_IN];                    // 32 KB
  int tid = threadIdx.x;
  const float* yl = Y + (size_t)(T_SZ - 1)*B_SZ*H_SZ;
  for (int o = tid; o < B_SZ*D_IN; o += 512){
    int b = o >> 8, d = o & 255;
    float s = 0.f;
    for (int h = 0; h < H_SZ; ++h) s += yl[b*H_SZ + h] * QW[h*D_IN + d];
    q0[o] = s;
  }
  __syncthreads();
  for (int o = tid; o < B_SZ*H_SZ; o += 512){
    int b = o >> 9, h = o & 511;
    float s = 0.f;
    for (int d = 0; d < D_IN; ++d) s += KW[h*D_IN + d] * q0[b*D_IN + d];
    qp[o] = s;
  }
}

// scores[b,t] = tanh(qp[b] . Y[t,b]) — one wave per score, lane-parallel over H
__global__ void k_scores(const float* __restrict__ Y, const float* __restrict__ qp,
                         float* __restrict__ sc){
  int lane = threadIdx.x & 31;
  int wid = blockIdx.x*(blockDim.x >> 5) + (threadIdx.x >> 5);
  int t = wid >> 5, b = wid & 31;
  const float* y = Y + ((size_t)t*B_SZ + b)*H_SZ;
  const float* q = qp + b*H_SZ;
  float s = 0.f;
  #pragma unroll
  for (int k = 0; k < 4; ++k){
    v4f yv = *(const v4f*)(y + k*128 + lane*4);
    v4f qv = *(const v4f*)(q + k*128 + lane*4);
    s += yv.x*qv.x + yv.y*qv.y + yv.z*qv.z + yv.w*qv.w;
  }
  #pragma unroll
  for (int m = 16; m >= 1; m >>= 1) s += __shfl_xor(s, m, 32);
  if (lane == 0) sc[b*T_SZ + t] = fast_tanh(s);
}

__global__ void k_softmax(const float* __restrict__ sc, float* __restrict__ att){
  __shared__ float red[256];
  int b = blockIdx.x, tid = threadIdx.x;
  const float* rowp = sc + b*T_SZ;
  float m = -1e30f;
  for (int t = tid; t < T_SZ; t += 256) m = fmaxf(m, rowp[t]);
  red[tid] = m; __syncthreads();
  for (int s = 128; s > 0; s >>= 1){ if (tid < s) red[tid] = fmaxf(red[tid], red[tid+s]); __syncthreads(); }
  m = red[0]; __syncthreads();
  float sum = 0.f;
  for (int t = tid; t < T_SZ; t += 256) sum += __expf(rowp[t] - m);
  red[tid] = sum; __syncthreads();
  for (int s = 128; s > 0; s >>= 1){ if (tid < s) red[tid] += red[tid+s]; __syncthreads(); }
  float inv = 1.0f / red[0];
  for (int t = tid; t < T_SZ; t += 256) att[b*T_SZ + t] = __expf(rowp[t] - m) * inv;
}

// c partials: 8 t-chunks per batch row (deterministic, no atomics)
__global__ void k_ctx1(const float* __restrict__ Y, const float* __restrict__ att,
                       float* __restrict__ cp){
  __shared__ float als[256];
  int b = blockIdx.x, ch = blockIdx.y, h = threadIdx.x;
  if (h < 256) als[h] = att[b*T_SZ + ch*256 + h];
  __syncthreads();
  float acc = 0.f;
  for (int tt = 0; tt < 256; ++tt){
    int t = ch*256 + tt;
    acc += als[tt] * Y[((size_t)t*B_SZ + b)*H_SZ + h];
  }
  cp[((size_t)ch*B_SZ + b)*H_SZ + h] = acc;
}

__global__ void k_ctx2(const float* __restrict__ cp, float* __restrict__ cvec){
  int o = blockIdx.x*256 + threadIdx.x;       // 32*512 outputs
  float s = 0.f;
  #pragma unroll
  for (int ch = 0; ch < 8; ++ch) s += cp[(size_t)ch*B_SZ*H_SZ + o];
  cvec[o] = s;
}

// ---------------------------------------------------------------------------
extern "C" void kernel_launch(void* const* d_in, const int* in_sizes, int n_in,
                              void* d_out, int out_size, void* d_ws, size_t ws_size,
                              hipStream_t stream){
  (void)in_sizes; (void)n_in; (void)out_size; (void)ws_size;
  const float* X   = (const float*)d_in[0];
  const float* Wxh = (const float*)d_in[1];
  const float* Whh = (const float*)d_in[2];
  const float* bh  = (const float*)d_in[3];
  const float* QW  = (const float*)d_in[4];
  const float* KW  = (const float*)d_in[5];
  float* out = (float*)d_out;
  char* ws = (char*)d_ws;
  _Float16* whh16 = (_Float16*)(ws + WS_WHH16);
  _Float16* wxh16 = (_Float16*)(ws + WS_WXH16);
  _Float16* xw16  = (_Float16*)(ws + WS_XW16);
  float* qp  = (float*)(ws + WS_QP);
  float* sc  = (float*)(ws + WS_SCORES);
  float* att = (float*)(ws + WS_ATT);
  float* cp  = (float*)(ws + WS_CPART);
  float* Y    = out;                                   // [T,B,H]
  float* cvec = out + (size_t)T_SZ*B_SZ*H_SZ;          // [B,H]

  k_prep   <<<1536, 256, 0,          stream>>>(Wxh, Whh, whh16, wxh16);
  k_xw     <<<256,  512, 256*1024,   stream>>>(X, bh, wxh16, xw16);
  k_scan   <<<2,    512, 256*1024,   stream>>>(whh16, xw16, Y);
  k_q0qp   <<<1,    512, 0,          stream>>>(Y, QW, KW, qp);
  k_scores <<<8192, 256, 0,          stream>>>(Y, qp, sc);
  k_softmax<<<32,   256, 0,          stream>>>(sc, att);
  k_ctx1   <<<dim3(32,8), 512, 0,    stream>>>(Y, att, cp);
  k_ctx2   <<<64,   256, 0,          stream>>>(cp, cvec);
}
